// MultiHeadAttention_38689065402805
// MI455X (gfx1250) — matrix operand order repaired and checked by reading, not compiled, not run on hardware
//
#include <hip/hip_runtime.h>
#include <hip/hip_bf16.h>

// ---------------------------------------------------------------------------
// MHA for gfx1250 (MI455X, CDNA5): wave32 + v_wmma_f32_16x16x32_f16 pipeline.
// Pipeline: [weight transpose f32->f16] -> [Q/K/V projection WMMA GEMMs]
//           -> [flash-attention WMMA kernel] -> [output projection WMMA GEMM]
// Workspace: 4 * 16MB f16 activations + 4 * 2MB f16 weights = ~72MB (< 192MB L2).
// GEMM waves compute 16x64 strips (A reused across 4 WMMAs) for ~10.7 FLOP/B.
// ---------------------------------------------------------------------------

typedef _Float16 h16;
typedef __attribute__((ext_vector_type(16))) _Float16 v16h;
typedef __attribute__((ext_vector_type(8)))  _Float16 v8h;
typedef __attribute__((ext_vector_type(8)))  float    v8f;
typedef __attribute__((ext_vector_type(4)))  float    v4f;
typedef __attribute__((ext_vector_type(4)))  unsigned int v4u;

static constexpr int Bc  = 4;
static constexpr int LQc = 2048;
static constexpr int LKc = 2048;
static constexpr int DM  = 1024;     // d_model
static constexpr int NH  = 16;
static constexpr int DKc = 64;       // per-head dim
static constexpr int Mtot = Bc * LQc;                  // 8192 rows
static constexpr size_t ACT_ELEMS = (size_t)Mtot * DM; // 8M halfs per activation
static constexpr size_t WT_ELEMS  = (size_t)DM * DM;   // 1M halfs per weight

static __device__ __forceinline__ v8f wmma16(v16h a, v16h b, v8f c) {
  // D = A(16x32 f16) x B(32x16 f16) + C(16x16 f32)
  return __builtin_amdgcn_wmma_f32_16x16x32_f16(
      /*neg_a=*/false, a, /*neg_b=*/false, b,
      /*c_mod=*/(short)0, c, /*reuse_a=*/false, /*reuse_b=*/false);
}

// ---------------------------------------------------------------------------
// Weight transpose + f32->f16 convert: Wt[n][k] = (f16) W[k][n]   (DM x DM)
// ---------------------------------------------------------------------------
__global__ void wt_kernel(const float* __restrict__ W, h16* __restrict__ Wt) {
  int idx = blockIdx.x * blockDim.x + threadIdx.x;   // over DM*DM
  int n = idx >> 10;
  int k = idx & (DM - 1);
  Wt[idx] = (h16)W[(size_t)k * DM + n];
}

// ---------------------------------------------------------------------------
// Projection GEMM: Y = X(f32, Mtot x DM) * W (via Wt f16, N x K layout).
// One wave computes a 16x64 strip: the converted A operand feeds 4 WMMAs.
// mode 0: store as [B,H,L,64] (Q,K).  mode 1: store transposed [B,H,64,L] (V).
// ---------------------------------------------------------------------------
__global__ void proj_kernel(const float* __restrict__ X, const h16* __restrict__ Wt,
                            h16* __restrict__ Y, int mode) {
  const int lane   = threadIdx.x & 31;
  const int l16    = lane & 15;
  const int halfid = lane >> 4;
  const int o0     = halfid * 8;
  const int wid    = blockIdx.x * 8 + (threadIdx.x >> 5);
  const int nstrips = DM / 64;                       // 16
  const int mbase  = (wid / nstrips) * 16;
  const int nbase  = (wid % nstrips) * 64;

  const float* xrow = X + (size_t)(mbase + l16) * DM;
  const h16* wrow[4];
  #pragma unroll
  for (int j = 0; j < 4; ++j)
    wrow[j] = Wt + (size_t)(nbase + j * 16 + l16) * DM + halfid * 16;

  v8f acc[4];
  #pragma unroll
  for (int j = 0; j < 4; ++j) acc[j] = (v8f){};

  #pragma unroll 2
  for (int c = 0; c < DM; c += 32) {
    // A operand (contiguous K): lane<16 -> K {c..c+15}, lane>=16 -> K {c+16..c+31}
    v4f f0 = *(const v4f*)(xrow + c + 2 * o0);
    v4f f1 = *(const v4f*)(xrow + c + 2 * o0 + 4);
    v4f f2 = *(const v4f*)(xrow + c + 2 * o0 + 8);
    v4f f3 = *(const v4f*)(xrow + c + 2 * o0 + 12);
    v16h a;
    #pragma unroll
    for (int i = 0; i < 4; ++i) {
      a[i]      = (h16)f0[i];
      a[4 + i]  = (h16)f1[i];
      a[8 + i]  = (h16)f2[i];
      a[12 + i] = (h16)f3[i];
    }
    // B operands: column N, 16 consecutive K values (32B contiguous per lane)
    #pragma unroll
    for (int j = 0; j < 4; ++j) {
      v16h bm = *(const v16h*)(wrow[j] + c);
      acc[j] = wmma16(a, bm, acc[j]);
    }
  }

  #pragma unroll
  for (int j = 0; j < 4; ++j) {
    #pragma unroll
    for (int r = 0; r < 8; ++r) {
      int m  = mbase + r + halfid * 8;
      int n  = nbase + j * 16 + l16;
      int bb = m >> 11, l  = m & (LQc - 1);
      int hh = n >> 6,  dk = n & (DKc - 1);
      size_t idx = (mode == 0)
          ? ((((size_t)bb * NH + hh) * LQc + l) * DKc + dk)
          : ((((size_t)bb * NH + hh) * DKc + dk) * LKc + l);
      Y[idx] = (h16)acc[j][r];
    }
  }
}

// ---------------------------------------------------------------------------
// Flash attention: one wave handles a 16-row q tile for one (b,h).
// Qh,Kh : [B,H,L,64] f16.  Vt : [B,H,64,L] f16.  Oa out: [Mtot, DM] f16.
// Softmax scale (1/sqrt(64)=0.125, exact power of two) folded into Q at load.
// Mask tile (16x32 bytes) loaded cooperatively as one b128 per lane, then
// redistributed through LDS (DS ops from one wave complete in order).
// ---------------------------------------------------------------------------
__global__ void attn_kernel(const h16* __restrict__ Qh, const h16* __restrict__ Kh,
                            const h16* __restrict__ Vt,
                            const unsigned char* __restrict__ mask,
                            h16* __restrict__ Oa) {
  __shared__ __align__(16) h16 plds[8 * 16 * 40];           // P staging, padded rows
  __shared__ __align__(16) unsigned char mlds[8 * 16 * 32]; // mask tiles
  const int lane   = threadIdx.x & 31;
  const int l16    = lane & 15;
  const int halfid = lane >> 4;
  const int o0     = halfid * 8;
  const int slot   = threadIdx.x >> 5;
  const int wid    = blockIdx.x * 8 + slot;
  const int qtiles = LQc / 16;                      // 128
  const int bh     = wid / qtiles;
  const int qbase  = (wid % qtiles) * 16;
  const int b      = bh / NH;
  const int hh     = bh % NH;

  // Q tile in A-layout, loaded once, pre-scaled by 0.125
  const h16* qrow = Qh + ((size_t)bh * LQc + qbase + l16) * DKc;
  v16h qa[2];
  #pragma unroll
  for (int dc = 0; dc < 2; ++dc) {
    v8h x0 = *(const v8h*)(qrow + dc * 32 + 2 * o0);
    v8h x1 = *(const v8h*)(qrow + dc * 32 + 2 * o0 + 8);
    #pragma unroll
    for (int i = 0; i < 8; ++i) {
      qa[dc][i]     = x0[i] * (h16)0.125f;
      qa[dc][8 + i] = x1[i] * (h16)0.125f;
    }
  }

  const h16* kb = Kh + (size_t)bh * LKc * DKc;
  const h16* vb = Vt + (size_t)bh * DKc * LKc;
  h16* pl = plds + slot * (16 * 40);
  unsigned char* ml = mlds + slot * (16 * 32);
  // cooperative mask-tile addressing: lane covers row lane>>1, 16-byte half (lane&1)
  const unsigned char* mtile_src =
      mask + ((size_t)b * LQc + qbase + (lane >> 1)) * (size_t)LKc + (lane & 1) * 16;
  unsigned char* mtile_dst = ml + (lane >> 1) * 32 + (lane & 1) * 16;

  float mrow[8], lrow[8];
  #pragma unroll
  for (int r = 0; r < 8; ++r) { mrow[r] = -3.0e38f; lrow[r] = 0.0f; }
  v8f oacc[4];
  #pragma unroll
  for (int d = 0; d < 4; ++d) oacc[d] = (v8f){};

  for (int kc = 0; kc < LKc; kc += 32) {
    if (kc + 32 < LKc)  // uniform branch; warm next K chunk -> global_prefetch_b8
      __builtin_prefetch(kb + (size_t)(kc + 32 + l16) * DKc, 0, 3);

    // cooperative mask tile load: one b128 per lane covers the 16x32 tile
    *(v4u*)mtile_dst = *(const v4u*)(mtile_src + kc);

    // S = Q * K^T for 32 keys: two 16x16 f32 tiles, contraction over d=64
    v8f s0 = {}, s1 = {};
    #pragma unroll
    for (int dc = 0; dc < 2; ++dc) {
      v16h k0 = *(const v16h*)(kb + (size_t)(kc + l16) * DKc + dc * 32 + halfid * 16);
      v16h k1 = *(const v16h*)(kb + (size_t)(kc + 16 + l16) * DKc + dc * 32 + halfid * 16);
      s0 = wmma16(qa[dc], k0, s0);
      s1 = wmma16(qa[dc], k1, s1);
    }

    // mask + online softmax (row M = r + halfid*8, col N = l16 per C layout)
    #pragma unroll
    for (int r = 0; r < 8; ++r) {
      int Mr = r + halfid * 8;
      float v0 = s0[r];
      float v1 = s1[r];
      if (!ml[Mr * 32 + l16])      v0 = -3.0e38f;
      if (!ml[Mr * 32 + l16 + 16]) v1 = -3.0e38f;
      float rm = fmaxf(v0, v1);
      #pragma unroll
      for (int off = 1; off < 16; off <<= 1) rm = fmaxf(rm, __shfl_xor(rm, off, 32));
      float mo = mrow[r];
      float mn = fmaxf(mo, rm);
      float corr = __expf(mo - mn);
      float p0 = __expf(v0 - mn);
      float p1 = __expf(v1 - mn);
      float rs = p0 + p1;
      #pragma unroll
      for (int off = 1; off < 16; off <<= 1) rs += __shfl_xor(rs, off, 32);
      mrow[r] = mn;
      lrow[r] = lrow[r] * corr + rs;
      #pragma unroll
      for (int d = 0; d < 4; ++d) oacc[d][r] *= corr;
      pl[Mr * 40 + l16]      = (h16)p0;             // stage P for C->A re-layout
      pl[Mr * 40 + l16 + 16] = (h16)p1;
    }

    // Read P back in A-layout (same-wave DS ops are in-order; no barrier needed)
    v8h pa0 = *(const v8h*)(pl + l16 * 40 + 2 * o0);
    v8h pa1 = *(const v8h*)(pl + l16 * 40 + 2 * o0 + 8);
    v16h pa;
    #pragma unroll
    for (int i = 0; i < 8; ++i) { pa[i] = pa0[i]; pa[8 + i] = pa1[i]; }

    // O += P(16x32) * V(32x64): 4 WMMAs over d chunks of 16; Vt layout gives
    // contiguous per-lane B loads (column d, 16 consecutive k).
    #pragma unroll
    for (int d = 0; d < 4; ++d) {
      v16h vt = *(const v16h*)(vb + (size_t)(d * 16 + l16) * LKc + kc + halfid * 16);
      oacc[d] = wmma16(pa, vt, oacc[d]);
    }
  }

  // normalize and store attention output [row = b*LQ+q, col = h*64+d]
  #pragma unroll
  for (int r = 0; r < 8; ++r) {
    float inv = 1.0f / lrow[r];
    size_t row = (size_t)b * LQc + qbase + r + halfid * 8;
    #pragma unroll
    for (int d = 0; d < 4; ++d)
      Oa[row * DM + hh * DKc + d * 16 + l16] = (h16)(oacc[d][r] * inv);
  }
}

// ---------------------------------------------------------------------------
// Output GEMM: out(f32) = Oa(f16, Mtot x DM) * Wo (via WoT f16, N x K).
// One wave computes a 16x64 strip (A reused across 4 WMMAs per k-chunk).
// ---------------------------------------------------------------------------
__global__ void out_kernel(const h16* __restrict__ A, const h16* __restrict__ WoT,
                           float* __restrict__ Y) {
  const int lane   = threadIdx.x & 31;
  const int l16    = lane & 15;
  const int halfid = lane >> 4;
  const int o0     = halfid * 8;
  const int wid    = blockIdx.x * 8 + (threadIdx.x >> 5);
  const int nstrips = DM / 64;                       // 16
  const int mbase  = (wid / nstrips) * 16;
  const int nbase  = (wid % nstrips) * 64;

  const h16* arow = A + (size_t)(mbase + l16) * DM;
  const h16* wrow[4];
  #pragma unroll
  for (int j = 0; j < 4; ++j)
    wrow[j] = WoT + (size_t)(nbase + j * 16 + l16) * DM + halfid * 16;

  v8f acc[4];
  #pragma unroll
  for (int j = 0; j < 4; ++j) acc[j] = (v8f){};

  #pragma unroll 2
  for (int c = 0; c < DM; c += 32) {
    v8h a0 = *(const v8h*)(arow + c + 2 * o0);
    v8h a1 = *(const v8h*)(arow + c + 2 * o0 + 8);
    v16h a;
    #pragma unroll
    for (int i = 0; i < 8; ++i) { a[i] = a0[i]; a[8 + i] = a1[i]; }
    #pragma unroll
    for (int j = 0; j < 4; ++j) {
      v16h bm = *(const v16h*)(wrow[j] + c);
      acc[j] = wmma16(a, bm, acc[j]);
    }
  }
  #pragma unroll
  for (int j = 0; j < 4; ++j)
    #pragma unroll
    for (int r = 0; r < 8; ++r)
      Y[(size_t)(mbase + r + halfid * 8) * DM + nbase + j * 16 + l16] = acc[j][r];
}

// ---------------------------------------------------------------------------
extern "C" void kernel_launch(void* const* d_in, const int* in_sizes, int n_in,
                              void* d_out, int out_size, void* d_ws, size_t ws_size,
                              hipStream_t stream) {
  const float* q   = (const float*)d_in[0];
  const float* k   = (const float*)d_in[1];
  const float* v   = (const float*)d_in[2];
  const unsigned char* mask = (const unsigned char*)d_in[3];
  const float* Wq  = (const float*)d_in[4];
  const float* Wk  = (const float*)d_in[5];
  const float* Wv  = (const float*)d_in[6];
  const float* Wo  = (const float*)d_in[7];
  float* out = (float*)d_out;

  // workspace carve-up (f16): 4 activations + 4 transposed weights (~72MB)
  h16* ws  = (h16*)d_ws;
  h16* Qh  = ws;
  h16* Kh  = ws + 1 * ACT_ELEMS;
  h16* Vt  = ws + 2 * ACT_ELEMS;
  h16* Oa  = ws + 3 * ACT_ELEMS;
  h16* WqT = ws + 4 * ACT_ELEMS;
  h16* WkT = WqT + WT_ELEMS;
  h16* WvT = WkT + WT_ELEMS;
  h16* WoT = WvT + WT_ELEMS;

  const int wtBlocks   = (DM * DM) / 256;                       // 4096
  const int gemmBlocks = ((Mtot / 16) * (DM / 64)) / 8;         // 1024 (8 waves/blk)
  const int attnBlocks = (Bc * NH * (LQc / 16)) / 8;            // 1024

  wt_kernel<<<wtBlocks, 256, 0, stream>>>(Wq, WqT);
  wt_kernel<<<wtBlocks, 256, 0, stream>>>(Wk, WkT);
  wt_kernel<<<wtBlocks, 256, 0, stream>>>(Wv, WvT);
  wt_kernel<<<wtBlocks, 256, 0, stream>>>(Wo, WoT);

  proj_kernel<<<gemmBlocks, 256, 0, stream>>>(q, WqT, Qh, 0);
  proj_kernel<<<gemmBlocks, 256, 0, stream>>>(k, WkT, Kh, 0);
  proj_kernel<<<gemmBlocks, 256, 0, stream>>>(v, WvT, Vt, 1);

  attn_kernel<<<attnBlocks, 256, 0, stream>>>(Qh, Kh, Vt, mask, Oa);

  out_kernel<<<gemmBlocks, 256, 0, stream>>>(Oa, WoT, out);
}
